// HierarchicalAttention_3384434229843
// MI455X (gfx1250) — matrix operand order repaired and checked by reading, not compiled, not run on hardware
//
#include <hip/hip_runtime.h>
#include <hip/hip_bf16.h>
#include <cstddef>

// Problem constants (fixed by the reference harness)
#define B_  4
#define T_  50
#define D_  512
#define SRC_ 400
#define SENTS_ 16
#define WORDS_ 40
#define NKEYS_ (SRC_ + SENTS_*WORDS_)   // 1040
#define NEGBIG (-1e30f)

typedef __attribute__((ext_vector_type(16))) __bf16 bf16x16;
typedef __attribute__((ext_vector_type(8)))  float  f32x8;

// ---------------------------------------------------------------------------
// WMMA K-loop core: 64(M) x 16(N) tile, B fragment shared by 4 WMMAs.
// MASKED=true multiplies A by a per-subtile validity mask (edge tile only).
// ---------------------------------------------------------------------------
template <bool MASKED>
__device__ __forceinline__ void gemm_core(
    const float* const* Ar, const float* __restrict__ W,
    int col, int hi, const float* zm, f32x8* acc)
{
#pragma unroll 2
    for (int k0 = 0; k0 < D_; k0 += 32) {
        // B 32x16 bf16 fragment: K = e + hi*16, N = col
        bf16x16 bfrag;
#pragma unroll
        for (int e = 0; e < 16; ++e)
            bfrag[e] = (__bf16)W[(size_t)(k0 + (hi << 4) + e) * D_ + col];
#pragma unroll
        for (int r = 0; r < 4; ++r) {
            // A 16x32 bf16 fragment: K = e + hi*16
            bf16x16 afrag;
#pragma unroll
            for (int e = 0; e < 16; ++e) {
                const int ka = k0 + (hi << 4) + e;
                float fv = Ar[r][ka];
                if (MASKED) fv *= zm[r];
                afrag[e] = (__bf16)fv;
            }
            acc[r] = __builtin_amdgcn_wmma_f32_16x16x32_bf16(
                /*neg_a=*/false, afrag, /*neg_b=*/false, bfrag,
                /*c_mod=*/(short)0, acc[r], /*reuse_a=*/false, /*reuse_b=*/false);
        }
    }
}

// ---------------------------------------------------------------------------
// GEMM: C[M,512] = A[M,512] @ W[512,512] (+ bias), fp32 in/out, bf16 WMMA core.
// Each wave owns a 64(M) x 16(N) tile; 8 waves per 256-thread block.
// ---------------------------------------------------------------------------
__global__ __launch_bounds__(256) void gemm512_bf16_wmma(
    const float* __restrict__ A, const float* __restrict__ W,
    const float* __restrict__ bias, float* __restrict__ C,
    int M, int tiles)
{
    const int lane = threadIdx.x & 31;
    const int wv   = threadIdx.x >> 5;
    const int tile = blockIdx.x * 8 + wv;
    if (tile >= tiles) return;

    const int mt = tile >> 5;        // 512/16 = 32 tiles along N
    const int nt = tile & 31;
    const int hi = lane >> 4;        // lane group 0/1
    const int lo = lane & 15;
    const int col   = nt * 16 + lo;  // C/B column this lane owns
    const int mbase = mt * 64;

    const float* Ar[4];
    float zm[4];
#pragma unroll
    for (int r = 0; r < 4; ++r) {
        const int row = mbase + r * 16 + lo;
        Ar[r] = A + (size_t)((row < M) ? row : 0) * D_;
        zm[r] = (row < M) ? 1.0f : 0.0f;
    }

    f32x8 acc[4];
#pragma unroll
    for (int r = 0; r < 4; ++r)
#pragma unroll
        for (int e = 0; e < 8; ++e) acc[r][e] = 0.0f;

    const float bv = bias ? bias[col] : 0.0f;

    if (mbase + 64 <= M) {
        // Common path: whole tile valid -> no mask multiplies, branch-free stores
        gemm_core<false>(Ar, W, col, hi, zm, acc);
#pragma unroll
        for (int r = 0; r < 4; ++r)
#pragma unroll
            for (int e = 0; e < 8; ++e) {
                const int m = mbase + r * 16 + (hi << 3) + e;  // C: M = e + hi*8
                C[(size_t)m * D_ + col] = acc[r][e] + bv;
            }
    } else {
        // Edge tile (only the last mt of M=200): masked A, guarded stores
        gemm_core<true>(Ar, W, col, hi, zm, acc);
#pragma unroll
        for (int r = 0; r < 4; ++r)
#pragma unroll
            for (int e = 0; e < 8; ++e) {
                const int m = mbase + r * 16 + (hi << 3) + e;
                if (m < M) C[(size_t)m * D_ + col] = acc[r][e] + bv;
            }
    }
}

// ---------------------------------------------------------------------------
// Additive-attention scores: out[bt, j] = v . tanh(wq[bt,:] + uh[row(j,b),:])
// One block per (b,t); each of 8 waves strides over keys; lanes split D.
// word_mode=0: row = j*B + b  (src_bank / qa_sent_bank native layout)
// word_mode=1: j = s*WORDS+w -> row = (w*B + b)*SENTS + s (qa_word_bank native)
// ---------------------------------------------------------------------------
__global__ __launch_bounds__(256) void score_kernel(
    const float* __restrict__ wq, const float* __restrict__ uh,
    const float* __restrict__ v, float* __restrict__ out,
    int nkeys, int word_mode)
{
    const int tid  = threadIdx.x;
    const int lane = tid & 31;
    const int wv   = tid >> 5;
    const int bt   = blockIdx.x;
    const int b    = bt / T_;

    const float* __restrict__ wqr = wq + (size_t)bt * D_;
    float wql[16], vl[16];
#pragma unroll
    for (int i = 0; i < 16; ++i) {
        wql[i] = wqr[lane + i * 32];
        vl[i]  = v[lane + i * 32];
    }

    for (int j = wv; j < nkeys; j += 8) {
        int row;
        if (word_mode) {
            const int w = j % WORDS_, s = j / WORDS_;
            row = (w * B_ + b) * SENTS_ + s;
        } else {
            row = j * B_ + b;
        }
        const float* __restrict__ uhr = uh + (size_t)row * D_;
        float s = 0.0f;
#pragma unroll
        for (int i = 0; i < 16; ++i)
            s += vl[i] * tanhf(wql[i] + uhr[lane + i * 32]);
#pragma unroll
        for (int off = 16; off; off >>= 1)
            s += __shfl_xor(s, off, 32);
        if (lane == 0) out[(size_t)bt * nkeys + j] = s;
    }
}

// ---------------------------------------------------------------------------
// Hierarchical combine + length mask + softmax + context GEMV.
// One block (256 threads) per (b,t); 1040 probs in LDS; each thread owns
// two D-columns of the output context vector.
// ---------------------------------------------------------------------------
__global__ __launch_bounds__(256) void softmax_ctx_kernel(
    const float* __restrict__ sc_pass, const float* __restrict__ sc_sent,
    const float* __restrict__ sc_word,
    const int* __restrict__ src_len, const int* __restrict__ word_len,
    const float* __restrict__ src_bank, const float* __restrict__ qa_word_bank,
    float* __restrict__ out)
{
    __shared__ float p[NKEYS_];
    __shared__ float red[8];
    const int tid = threadIdx.x;
    const int bt  = blockIdx.x;
    const int b   = bt / T_;
    const int slen = src_len[b];

    // combine + mask
    for (int i = tid; i < NKEYS_; i += 256) {
        float val; bool ok;
        if (i < SRC_) {
            val = sc_pass[(size_t)bt * SRC_ + i];
            ok  = (i < slen);
        } else {
            const int j = i - SRC_;
            const int s = j / WORDS_, w = j - s * WORDS_;
            val = sc_word[(size_t)bt * (SENTS_*WORDS_) + j]
                * sc_sent[(size_t)bt * SENTS_ + s];
            ok  = (w < word_len[b * SENTS_ + s]);
        }
        p[i] = ok ? val : NEGBIG;
    }
    __syncthreads();

    // max
    float m = NEGBIG;
    for (int i = tid; i < NKEYS_; i += 256) m = fmaxf(m, p[i]);
#pragma unroll
    for (int off = 16; off; off >>= 1) m = fmaxf(m, __shfl_xor(m, off, 32));
    if ((tid & 31) == 0) red[tid >> 5] = m;
    __syncthreads();
    float mx = red[0];
#pragma unroll
    for (int k = 1; k < 8; ++k) mx = fmaxf(mx, red[k]);
    __syncthreads();

    // exp + sum
    float ssum = 0.0f;
    for (int i = tid; i < NKEYS_; i += 256) {
        const float e = __expf(p[i] - mx);
        p[i] = e;
        ssum += e;
    }
#pragma unroll
    for (int off = 16; off; off >>= 1) ssum += __shfl_xor(ssum, off, 32);
    if ((tid & 31) == 0) red[tid >> 5] = ssum;
    __syncthreads();
    float tot = 0.0f;
#pragma unroll
    for (int k = 0; k < 8; ++k) tot += red[k];
    const float inv = 1.0f / tot;
    for (int i = tid; i < NKEYS_; i += 256) p[i] *= inv;
    __syncthreads();

    // context: c[d] = sum_i p[i] * mem[i, d]; thread owns d = tid, tid+256
    float acc0 = 0.0f, acc1 = 0.0f;
    for (int i = 0; i < NKEYS_; ++i) {
        const float pi = p[i];
        if (pi == 0.0f) continue;               // uniform across the block
        const float* __restrict__ mrow;
        if (i < SRC_) {
            mrow = src_bank + ((size_t)(i * B_ + b)) * D_;
        } else {
            const int j = i - SRC_;
            const int s = j / WORDS_, w = j - s * WORDS_;
            mrow = qa_word_bank + ((size_t)((w * B_ + b) * SENTS_ + s)) * D_;
        }
        acc0 = fmaf(pi, mrow[tid],       acc0);
        acc1 = fmaf(pi, mrow[tid + 256], acc1);
    }
    out[(size_t)bt * D_ + tid]       = acc0;
    out[(size_t)bt * D_ + 256 + tid] = acc1;
}

// ---------------------------------------------------------------------------
// Host-side launcher (graph-capture safe: launches only, all on `stream`)
// ---------------------------------------------------------------------------
extern "C" void kernel_launch(void* const* d_in, const int* in_sizes, int n_in,
                              void* d_out, int out_size, void* d_ws, size_t ws_size,
                              hipStream_t stream) {
    (void)in_sizes; (void)n_in; (void)out_size; (void)ws_size;

    const float* source        = (const float*)d_in[0];   // [B,T,D]
    const float* src_bank      = (const float*)d_in[1];   // [SRC,B,D]
    const int*   src_lengths   = (const int*)  d_in[2];   // [B]
    const float* qa_sent_bank  = (const float*)d_in[3];   // [SENTS,B,D]
    /* d_in[4] = qa_sent_lengths: unused, as in reference */
    const float* qa_word_bank  = (const float*)d_in[5];   // [WORDS,B,SENTS,D]
    const int*   qa_word_len   = (const int*)  d_in[6];   // [B,SENTS]
    const float* Wq_word = (const float*)d_in[7];
    const float* bq_word = (const float*)d_in[8];
    const float* Uc_word = (const float*)d_in[9];
    const float* v_word  = (const float*)d_in[10];
    const float* Wq_sent = (const float*)d_in[11];
    const float* bq_sent = (const float*)d_in[12];
    const float* Uc_sent = (const float*)d_in[13];
    const float* v_sent  = (const float*)d_in[14];
    const float* Wq_pass = (const float*)d_in[15];
    const float* bq_pass = (const float*)d_in[16];
    const float* Uc_pass = (const float*)d_in[17];
    const float* v_pass  = (const float*)d_in[18];

    float* ws = (float*)d_ws;
    // fp32 scratch layout (element offsets)
    const size_t M_Q    = B_ * T_;                      // 200
    const size_t M_WORD = WORDS_ * B_ * SENTS_;         // 2560
    const size_t M_PASS = SRC_ * B_;                    // 1600
    const size_t M_SENT = SENTS_ * B_;                  // 64
    float* wq_word = ws;                    // 200*512
    float* wq_sent = wq_word + M_Q * D_;
    float* wq_pass = wq_sent + M_Q * D_;
    float* uh_word = wq_pass + M_Q * D_;    // 2560*512
    float* uh_pass = uh_word + M_WORD * D_; // 1600*512
    float* uh_sent = uh_pass + M_PASS * D_; // 64*512
    float* sc_pass = uh_sent + M_SENT * D_; // 200*400
    float* sc_sent = sc_pass + M_Q * SRC_;  // 200*16
    float* sc_word = sc_sent + M_Q * SENTS_;// 200*640

    auto launch_gemm = [&](const float* A, const float* W, const float* bias,
                           float* C, int M) {
        const int tiles  = ((M + 63) / 64) * (D_ / 16);   // 64-row tiles x 32 col tiles
        const int blocks = (tiles + 7) / 8;
        gemm512_bf16_wmma<<<blocks, 256, 0, stream>>>(A, W, bias, C, M, tiles);
    };

    // Six GEMMs on the WMMA pipe
    launch_gemm(source,       Wq_word, bq_word, wq_word, (int)M_Q);
    launch_gemm(source,       Wq_sent, bq_sent, wq_sent, (int)M_Q);
    launch_gemm(source,       Wq_pass, bq_pass, wq_pass, (int)M_Q);
    launch_gemm(qa_word_bank, Uc_word, nullptr, uh_word, (int)M_WORD);
    launch_gemm(src_bank,     Uc_pass, nullptr, uh_pass, (int)M_PASS);
    launch_gemm(qa_sent_bank, Uc_sent, nullptr, uh_sent, (int)M_SENT);

    // tanh-score stage (transcendental VALU bound)
    score_kernel<<<B_ * T_, 256, 0, stream>>>(wq_pass, uh_pass, v_pass, sc_pass, SRC_, 0);
    score_kernel<<<B_ * T_, 256, 0, stream>>>(wq_sent, uh_sent, v_sent, sc_sent, SENTS_, 0);
    score_kernel<<<B_ * T_, 256, 0, stream>>>(wq_word, uh_word, v_word, sc_word, SENTS_ * WORDS_, 1);

    // combine + masked softmax + context
    softmax_ctx_kernel<<<B_ * T_, 256, 0, stream>>>(
        sc_pass, sc_sent, sc_word, src_lengths, qa_word_len,
        src_bank, qa_word_bank, (float*)d_out);
}
